// sparse_MLP_3375844295298
// MI455X (gfx1250) — matrix operand order repaired and checked by reading, not compiled, not run on hardware
//
#include <hip/hip_runtime.h>
#include <hip/hip_fp16.h>

// ---------------------------------------------------------------------------
// Autoregressive sparse-MLP flow for MI455X (gfx1250, wave32, WMMA).
// B=131072 rows, D=64 coords, HID=128. Each row independent -> the 64-step
// recurrence runs inside one workgroup; all three matmuls (64->128, 128->128,
// 128->1) run on v_wmma_f32_16x16x32_f16. Weights are double-buffered in LDS
// (stage idx+1 while computing idx); outputs stream straight to HBM.
// ---------------------------------------------------------------------------

typedef __attribute__((ext_vector_type(16))) _Float16 v16h;
typedef __attribute__((ext_vector_type(8)))  _Float16 v8h;
typedef __attribute__((ext_vector_type(4)))  _Float16 v4h;
typedef __attribute__((ext_vector_type(8)))  float    v8f;

#define D_DIM 64
#define HID   128
#define ROWS_PER_BLOCK 128   // 8 waves x 16 rows
#define THREADS 256

// LDS partition (bytes):
//   sW1[2]  : 2 * 128*64  f16 = 32768   @ 0
//   sW2[2]  : 2 * 128*128 f16 = 65536   @ 32768
//   sWfh[2] : 2 * 128     f16 =   512   @ 98304
//   per wave (8x): sInp 16*64 f16 (2048) | sH 16*128 f16 (4096)  @ 98816
#define LDS_BYTES (32768 + 65536 + 512 + 8 * (2048 + 4096))   // 147968 (~144.5 KB)

// Branch-free GELU (tanh form expressed as sigmoid):
//   gelu(x) ~= x * rcp(1 + exp2(x * fma(x^2, C1, C0)))
//   C0 = -2*log2(e)*0.7978845608, C1 = C0*0.044715
__device__ __forceinline__ float gelu_fast(float x) {
    float x2 = x * x;
    float q  = x * __builtin_fmaf(x2, -0.1029433f, -2.3022086f);
    float e  = __builtin_amdgcn_exp2f(q);
    return x * __builtin_amdgcn_rcpf(1.0f + e);
}

// A-fragment (16x32 f16, M=rows): lanes 0-15 row M=lane, elems 0..15 -> K=kc+0..15;
// lanes 16-31 carry K=kc+16..31 (matches ld_fragB).
__device__ __forceinline__ v16h ld_fragA(const _Float16* s, int stride, int kc,
                                         int m, int g) {
    union { v16h v; v8h h[2]; } u;
    const _Float16* p = s + m * stride + kc + g * 16;
    u.h[0] = *(const v8h*)(p);
    u.h[1] = *(const v8h*)(p + 8);
    return u.v;
}

// B-fragment (32x16 f16, K x N): lane holds N = n (caller passes n0+lane%16);
// lanes 0-15 carry K=kc+0..15, lanes 16-31 carry K=kc+16..31.
__device__ __forceinline__ v16h ld_fragB(const _Float16* s, int stride, int kc,
                                         int n, int g) {
    union { v16h v; v8h h[2]; } u;
    const _Float16* p = s + n * stride + kc + g * 16;
    u.h[0] = *(const v8h*)(p);
    u.h[1] = *(const v8h*)(p + 8);
    return u.v;
}

// Cooperative f32->f16 staging of step j's weights into LDS buffer `buf`.
__device__ __forceinline__ void stage_weights(const float* __restrict__ W1,
                                              const float* __restrict__ W2,
                                              const float* __restrict__ Wf,
                                              _Float16* sW1, _Float16* sW2,
                                              _Float16* sWfh, int j, int buf,
                                              int tid) {
    _Float16* dW1 = sW1  + buf * (HID * D_DIM);
    _Float16* dW2 = sW2  + buf * (HID * HID);
    _Float16* dWf = sWfh + buf * HID;

    const float4* g1 = (const float4*)(W1 + (size_t)j * HID * D_DIM);
#pragma unroll
    for (int i = 0; i < (HID * D_DIM) / 4 / THREADS; ++i) {      // 8 iters
        const int e = tid + i * THREADS;
        float4 v = g1[e];
        v4h h; h[0] = (_Float16)v.x; h[1] = (_Float16)v.y;
               h[2] = (_Float16)v.z; h[3] = (_Float16)v.w;
        ((v4h*)dW1)[e] = h;
    }
    const float4* g2 = (const float4*)(W2 + (size_t)j * HID * HID);
#pragma unroll
    for (int i = 0; i < (HID * HID) / 4 / THREADS; ++i) {        // 16 iters
        const int e = tid + i * THREADS;
        float4 v = g2[e];
        v4h h; h[0] = (_Float16)v.x; h[1] = (_Float16)v.y;
               h[2] = (_Float16)v.z; h[3] = (_Float16)v.w;
        ((v4h*)dW2)[e] = h;
    }
    if (tid < HID) dWf[tid] = (_Float16)Wf[(size_t)j * HID + tid];

    if (j + 1 < D_DIM) {   // warm near caches for the step after this one
        __builtin_prefetch((const char*)(W2 + (size_t)(j + 1) * HID * HID) + tid * 256, 0, 3);
        __builtin_prefetch((const char*)(W1 + (size_t)(j + 1) * HID * D_DIM) + tid * 128, 0, 3);
    }
}

__global__ __launch_bounds__(THREADS)
void armlp_flow_kernel(const float* __restrict__ z,  const float* __restrict__ W1,
                       const float* __restrict__ W2, const float* __restrict__ Wf,
                       float* __restrict__ out) {
    extern __shared__ char smem[];
    _Float16* sW1  = (_Float16*)(smem);              // 2 x [128][64]  (n-major)
    _Float16* sW2  = (_Float16*)(smem + 32768);      // 2 x [128][128]
    _Float16* sWfh = (_Float16*)(smem + 98304);      // 2 x [128]
    char*     per  = smem + 98816;

    const int tid  = threadIdx.x;
    const int wave = tid >> 5;
    const int lane = tid & 31;
    const int g    = (lane >> 4) & 1;                // lane half
    const int mr   = lane & 15;                      // row (A) or N column (B/C)

    char* wb = per + wave * (2048 + 4096);
    _Float16* sInp = (_Float16*)(wb);                // [16][64]  f16 merged input
    _Float16* sH   = (_Float16*)(wb + 2048);         // [16][128] f16 hidden staging

    const int baseRow = blockIdx.x * ROWS_PER_BLOCK + wave * 16;

    // ---- init: sInp = f16(z tile); stage step 0 weights into buffer 0 -----
    {
        const float4* gz = (const float4*)(z + (size_t)baseRow * D_DIM);
        for (int i = lane; i < 256; i += 32) {       // 16 rows * 16 float4
            float4 v = gz[i];
            v4h h; h[0] = (_Float16)v.x; h[1] = (_Float16)v.y;
                   h[2] = (_Float16)v.z; h[3] = (_Float16)v.w;
            ((v4h*)sInp)[i] = h;
        }
    }
    stage_weights(W1, W2, Wf, sW1, sW2, sWfh, 0, 0, tid);

#pragma unroll 2
    for (int idx = 0; idx < D_DIM; ++idx) {
        const int cur = idx & 1;
        // Barrier: (a) weights[idx] in buf[cur] are visible to all waves,
        //          (b) every wave finished step idx-1, which was the last
        //              reader of buf[cur^1] -> safe to overwrite it now.
        __syncthreads();

        if (idx + 1 < D_DIM)   // overlap: stage next step's weights during compute
            stage_weights(W1, W2, Wf, sW1, sW2, sWfh, idx + 1, cur ^ 1, tid);

        const _Float16* w1 = sW1  + cur * (HID * D_DIM);
        const _Float16* w2 = sW2  + cur * (HID * HID);
        const _Float16* wf = sWfh + cur * HID;

        // ---- layer 1: gelu( inp[16x64] @ W1[idx]^T ) -> sH[16x128] --------
        {
            v16h a0 = ld_fragA(sInp, D_DIM, 0,  mr, g);
            v16h a1 = ld_fragA(sInp, D_DIM, 32, mr, g);
#pragma unroll
            for (int nt = 0; nt < 8; ++nt) {
                const int n0 = nt * 16;
                v8f c = {};
                v16h b0 = ld_fragB(w1, D_DIM, 0,  n0 + mr, g);
                c = __builtin_amdgcn_wmma_f32_16x16x32_f16(false, a0, false, b0,
                                                           (short)0, c, false, false);
                v16h b1 = ld_fragB(w1, D_DIM, 32, n0 + mr, g);
                c = __builtin_amdgcn_wmma_f32_16x16x32_f16(false, a1, false, b1,
                                                           (short)0, c, false, false);
#pragma unroll
                for (int r = 0; r < 8; ++r) {
                    float gv = gelu_fast(c[r]);      // C layout: M=r+8*g, N=n0+mr
                    sH[(r + 8 * g) * HID + n0 + mr] = (_Float16)gv;
                }
            }
        }
        asm volatile("s_wait_dscnt 0" ::: "memory");

        // ---- layer 2: gelu( h[16x128] @ W2[idx]^T ) -> sH[16x128] ---------
        {
            v16h a[4];
#pragma unroll
            for (int kc = 0; kc < 4; ++kc)           // hoist A before sH rewrite
                a[kc] = ld_fragA(sH, HID, kc * 32, mr, g);
#pragma unroll
            for (int nt = 0; nt < 8; ++nt) {
                const int n0 = nt * 16;
                v8f c = {};
#pragma unroll
                for (int kc = 0; kc < 4; ++kc) {
                    v16h b = ld_fragB(w2, HID, kc * 32, n0 + mr, g);
                    c = __builtin_amdgcn_wmma_f32_16x16x32_f16(false, a[kc], false, b,
                                                               (short)0, c, false, false);
                }
#pragma unroll
                for (int r = 0; r < 8; ++r) {
                    float gv = gelu_fast(c[r]);
                    sH[(r + 8 * g) * HID + n0 + mr] = (_Float16)gv;
                }
            }
        }
        asm volatile("s_wait_dscnt 0" ::: "memory");

        // ---- final layer on WMMA: out = gelu( h . wf ) --------------------
        // B-fragment: N=0 column holds wf[K], all other N columns zero.
        {
            v8f c = {};
#pragma unroll
            for (int kc = 0; kc < 4; ++kc) {
                v16h a = ld_fragA(sH, HID, kc * 32, mr, g);
                union { v16h v; v8h h[2]; } u;
                const _Float16* p = wf + kc * 32 + g * 16;
                u.h[0] = *(const v8h*)(p);           // broadcast load, masked below
                u.h[1] = *(const v8h*)(p + 8);
                v16h bf = (mr == 0) ? u.v : (v16h){};
                c = __builtin_amdgcn_wmma_f32_16x16x32_f16(false, a, false, bf,
                                                           (short)0, c, false, false);
            }
            // Result rows live in N=0 lanes: lane 0 -> rows 0..7, lane 16 -> rows 8..15.
#pragma unroll
            for (int r = 0; r < 8; ++r) {
                float o = gelu_fast(c[r]);           // uniform compute, masked store
                if (mr == 0) {
                    const int row = r + 8 * g;
                    out[(size_t)(baseRow + row) * D_DIM + idx] = o;  // stream to HBM
                    sInp[row * D_DIM + idx] = (_Float16)o;  // becomes x for col < idx
                }
            }
            asm volatile("s_wait_dscnt 0" ::: "memory");
        }
    }
}

extern "C" void kernel_launch(void* const* d_in, const int* in_sizes, int n_in,
                              void* d_out, int out_size, void* d_ws, size_t ws_size,
                              hipStream_t stream) {
    (void)n_in; (void)out_size; (void)d_ws; (void)ws_size;
    const float* z  = (const float*)d_in[0];   // [B,64]
    const float* W1 = (const float*)d_in[1];   // [64,128,64]
    const float* W2 = (const float*)d_in[2];   // [64,1,128,128]
    const float* Wf = (const float*)d_in[3];   // [64,1,128]
    float* out = (float*)d_out;                // [B,64]

    const int rows = in_sizes[0] / D_DIM;      // 131072
    const int grid = rows / ROWS_PER_BLOCK;    // 1024
    hipLaunchKernelGGL(armlp_flow_kernel, dim3(grid), dim3(THREADS),
                       LDS_BYTES, stream, z, W1, W2, Wf, out);
}
